// MultiHeadAttention_644245095061
// MI455X (gfx1250) — matrix operand order repaired and checked by reading, not compiled, not run on hardware
//
#include <hip/hip_runtime.h>

#define B_  8
#define S_  1024
#define DM  1024
#define H_  16
#define DK  64
#define M_  (B_ * S_)

typedef __attribute__((ext_vector_type(16))) __bf16 v16bf;
typedef __attribute__((ext_vector_type(8)))  __bf16 v8bf;
typedef __attribute__((ext_vector_type(4)))  __bf16 v4bf;
typedef __attribute__((ext_vector_type(8)))  float  v8f;
typedef __attribute__((ext_vector_type(4)))  unsigned int u32x4;
typedef __attribute__((ext_vector_type(8)))  int i32x8;
typedef __attribute__((ext_vector_type(4)))  int i32x4;

static __device__ __forceinline__ v16bf cat16(v8bf lo, v8bf hi) {
  return __builtin_shufflevector(lo, hi, 0,1,2,3,4,5,6,7,8,9,10,11,12,13,14,15);
}
static __device__ __forceinline__ v8f zero8() {
  v8f z;
#pragma unroll
  for (int i = 0; i < 8; ++i) z[i] = 0.0f;
  return z;
}

// ----- raw v_max (skip llvm.maxnum canonicalization; WMMA outputs are
// already canonical per ISA: RNE + NaN propagation) --------------------------
static __device__ __forceinline__ float fmax_raw(float a, float b) {
  float d;
  asm("v_max_num_f32 %0, %1, %2" : "=v"(d) : "v"(a), "v"(b));
  return d;
}

// ----- fused DPP row_ror reductions: one VALU op per butterfly step ---------
// d = op(row_ror:N(s), s)  within each 16-lane row
#define F32_MAX_ROR(dst, src, N)                                         \
  asm("v_max_num_f32_dpp %0, %1, %1 row_ror:" #N                         \
      " row_mask:0xf bank_mask:0xf bound_ctrl:1"                         \
      : "=v"(dst) : "v"(src))
#define F32_ADD_ROR(dst, src, N)                                         \
  asm("v_add_f32_dpp %0, %1, %1 row_ror:" #N                             \
      " row_mask:0xf bank_mask:0xf bound_ctrl:1"                         \
      : "=v"(dst) : "v"(src))

static __device__ __forceinline__ float rowmax16(float x) {
  float t;
  F32_MAX_ROR(t, x, 1);
  F32_MAX_ROR(x, t, 2);
  F32_MAX_ROR(t, x, 4);
  F32_MAX_ROR(x, t, 8);
  return x;
}
static __device__ __forceinline__ float rowsum16(float x) {
  float t;
  F32_ADD_ROR(t, x, 1);
  F32_ADD_ROR(x, t, 2);
  F32_ADD_ROR(t, x, 4);
  F32_ADD_ROR(x, t, 8);
  return x;
}

// raw v_exp_f32 (underflow->0 is fine for softmax; skip libm denorm fixup)
static __device__ __forceinline__ float exp2_raw(float x) {
  return __builtin_amdgcn_exp2f(x);
}

// ----- Tensor Data Mover: 2D strided tile -> LDS (TENSORcnt tracked) --------
// D# per CDNA5 ISA. data_size=2B, 2D (groups 2/3 zero).
static __device__ __forceinline__ unsigned lds_off32(const void* p) {
  return (unsigned)(unsigned long long)p;   // low 32 bits = LDS byte offset
}
static __device__ __forceinline__ void tdm_load_2d(
    unsigned lds_addr, const void* gptr,
    unsigned tile_w, unsigned tile_h, unsigned stride_elems)
{
  const unsigned long long ga = (unsigned long long)gptr;
  u32x4 g0 = {
    1u,                                                 // count=1, user mode
    lds_addr,                                           // lds_addr
    (unsigned)(ga & 0xFFFFFFFFu),                       // global_addr[31:0]
    (unsigned)((ga >> 32) & 0x1FFFFFFu) | (2u << 30)    // ga[56:32] | type=2
  };
  i32x8 g1 = {
    (int)(1u << 16),                                    // data_size=1 -> 2B
    (int)((tile_w & 0xFFFFu) << 16),                    // tensor_dim0[15:0]
    (int)(((tile_w >> 16) & 0xFFFFu) |
          ((tile_h & 0xFFFFu) << 16)),                  // dim0[31:16]|dim1[15:0]
    (int)(((tile_h >> 16) & 0xFFFFu) |
          ((tile_w & 0xFFFFu) << 16)),                  // dim1[31:16]|tile_dim0
    (int)(tile_h & 0xFFFFu),                            // tile_dim1 (tile_dim2=0)
    (int)stride_elems,                                  // dim0_stride[31:0]
    0, 0                                                // strides hi (unused 2D)
  };
  i32x4 z4 = {0, 0, 0, 0};
#if defined(__clang_major__) && (__clang_major__ >= 23)
  i32x8 z8 = {0, 0, 0, 0, 0, 0, 0, 0};
  __builtin_amdgcn_tensor_load_to_lds(g0, g1, z4, z4, z8, 0);
#else
  __builtin_amdgcn_tensor_load_to_lds(g0, g1, z4, z4, 0);
#endif
}

// ---------------------------------------------------------------------------
// fp32 -> bf16 pack (weights + activations, once)
// ---------------------------------------------------------------------------
__global__ __launch_bounds__(256)
void cvt_f32_to_bf16(const float* __restrict__ in, __bf16* __restrict__ out, int n4) {
  const int i = blockIdx.x * 256 + threadIdx.x;
  if (i < n4) {
    float4 f = ((const float4*)in)[i];
    v4bf o;
    o[0] = (__bf16)f.x; o[1] = (__bf16)f.y; o[2] = (__bf16)f.z; o[3] = (__bf16)f.w;
    ((v4bf*)out)[i] = o;
  }
}

// ---------------------------------------------------------------------------
// WMMA GEMM: Y[m][n] = (sum_k A[m][k] * W[n][k] + bias[n]) * oscale
//  A: row-major [M_][DM] bf16;  W: row-major [DM][DM] bf16 (B = W^T)
//  OUT_MODE: 0 = bf16 natural, 1 = bf16 V^T layout out[(b*DM+n)*S_+s], 2 = f32
// ---------------------------------------------------------------------------
template<int OUT_MODE>
__global__ __launch_bounds__(128)
void gemm16(const __bf16* __restrict__ Ap, const __bf16* __restrict__ Bw,
            const float* __restrict__ bias, void* __restrict__ Out, float oscale)
{
  const int lane = threadIdx.x & 31;
  const int wave = threadIdx.x >> 5;
  const int half = lane >> 4;
  const int l15  = lane & 15;
  const int m0 = blockIdx.x * 64 + (wave & 1) * 32;
  const int n0 = blockIdx.y * 64 + (wave >> 1) * 32;

  v8f c[2][2];
  c[0][0] = zero8(); c[0][1] = zero8(); c[1][0] = zero8(); c[1][1] = zero8();

  for (int kb = 0; kb < DM; kb += 32) {
    v16bf a[2], b[2];
#pragma unroll
    for (int mi = 0; mi < 2; ++mi) {
      const __bf16* ap = Ap + (m0 + mi * 16 + l15) * DM + kb + half * 8;
      a[mi] = cat16(*(const v8bf*)ap, *(const v8bf*)(ap + 16));
    }
#pragma unroll
    for (int nj = 0; nj < 2; ++nj) {
      const __bf16* bp = Bw + (n0 + nj * 16 + l15) * DM + kb + half * 8;
      b[nj] = cat16(*(const v8bf*)bp, *(const v8bf*)(bp + 16));
    }
#pragma unroll
    for (int mi = 0; mi < 2; ++mi)
#pragma unroll
      for (int nj = 0; nj < 2; ++nj)
        c[mi][nj] = __builtin_amdgcn_wmma_f32_16x16x32_bf16(
            false, a[mi], false, b[nj], (short)0, c[mi][nj], false, false);
  }

#pragma unroll
  for (int mi = 0; mi < 2; ++mi)
#pragma unroll
    for (int nj = 0; nj < 2; ++nj)
#pragma unroll
      for (int i = 0; i < 8; ++i) {
        const int m = m0 + mi * 16 + half * 8 + i;
        const int n = n0 + nj * 16 + l15;
        const float v = (c[mi][nj][i] + bias[n]) * oscale;
        if (OUT_MODE == 2) {
          ((float*)Out)[m * DM + n] = v;
        } else if (OUT_MODE == 0) {
          ((__bf16*)Out)[m * DM + n] = (__bf16)v;
        } else {
          const int bb = m >> 10;                 // S_ == 1024
          const int ss = m & (S_ - 1);
          ((__bf16*)Out)[(bb * DM + n) * S_ + ss] = (__bf16)v;
        }
      }
}

// ---------------------------------------------------------------------------
// Flash attention. Block = 4 waves = 64 queries of one (b,h).
// K/V tiles (32 keys) staged into LDS by the Tensor Data Mover (wave 0 issues
// double-buffered tensor_load_to_lds; s_wait_tensorcnt + barrier publishes).
// Q pre-scaled by 0.125*log2(e) -> softmax in exp2 domain; row stats via
// fused DPP row_ror ops (pure VALU, 1 instr/step).
//  Qp, Kp : bf16 [B, S, DM]   Vt : bf16 [B, DM, S]   Ctx : bf16 [B, S, DM]
// ---------------------------------------------------------------------------
__global__ __launch_bounds__(128)
void flash_attn(const __bf16* __restrict__ Qp, const __bf16* __restrict__ Kp,
                const __bf16* __restrict__ Vt, __bf16* __restrict__ Ctx)
{
  __shared__ __align__(16) __bf16 lds_k[2][32][64];   // 8 KB: [buf][key][d]
  __shared__ __align__(16) __bf16 lds_v[2][64][32];   // 8 KB: [buf][d][s]
  __shared__ __align__(16) __bf16 lds_p[4][16 * 32];  // 4 KB: per-wave P tile

  const int tid  = threadIdx.x;
  const int lane = tid & 31;
  const int wave = tid >> 5;
  const int half = lane >> 4;
  const int l15  = lane & 15;
  const int qblk = blockIdx.x & 15;        // S_/64 = 16
  const int bh   = blockIdx.x >> 4;
  const int b    = bh >> 4;                // H_ = 16
  const int h    = bh & 15;
  const int q0   = qblk * 64 + wave * 16;

  // provably wave-uniform scalar guard (SCC branch, not EXEC predication):
  // TDM ops ignore EXEC, so they must be branched around, not predicated.
  const bool is_wave0 = (__builtin_amdgcn_readfirstlane(wave) == 0);

  const __bf16* Qb = Qp + b * S_ * DM + h * DK;
  const __bf16* Kb = Kp + b * S_ * DM + h * DK;
  const __bf16* Vb = Vt + (b * DM + h * DK) * S_;
  __bf16*       Cb = Ctx + b * S_ * DM + h * DK;

  // Q A-fragments (Q already carries 0.125*log2e)
  v16bf A[2];
#pragma unroll
  for (int dt = 0; dt < 2; ++dt) {
    const __bf16* qp = Qb + (q0 + l15) * DM + dt * 32 + half * 8;
    A[dt] = cat16(*(const v8bf*)qp, *(const v8bf*)(qp + 16));
  }

  v8f o[4];
#pragma unroll
  for (int f = 0; f < 4; ++f) o[f] = zero8();
  v8f mrow, lrow;
#pragma unroll
  for (int i = 0; i < 8; ++i) { mrow[i] = -3.0e38f; lrow[i] = 0.0f; }

  // one TDM descriptor pair per 32-key tile (issued by wave 0 only)
  auto stage = [&](int kb, int buf) {
    tdm_load_2d(lds_off32(&lds_k[buf][0][0]), Kb + kb * DM, DK, 32, DM);
    tdm_load_2d(lds_off32(&lds_v[buf][0][0]), Vb + kb,      32, DK, S_);
  };

  if (is_wave0) stage(0, 0);
  int cur = 0;

  for (int kb = 0; kb < S_; kb += 32) {
    const bool more = (kb + 32) < S_;
    if (is_wave0) {
      if (more) {
        stage(kb + 32, cur ^ 1);                     // prefetch next tile
        __builtin_amdgcn_s_wait_tensorcnt(2);        // current tile done
      } else {
        __builtin_amdgcn_s_wait_tensorcnt(0);
      }
    }
    __syncthreads();                                 // publish tile to block

    // scores S(16x32) = Qs . K^T  (already in exp2 domain)
    v16bf Bk[2][2];
#pragma unroll
    for (int kt = 0; kt < 2; ++kt)
#pragma unroll
      for (int dt = 0; dt < 2; ++dt) {
        const __bf16* kp = &lds_k[cur][kt * 16 + l15][dt * 32 + half * 8];
        Bk[kt][dt] = cat16(*(const v8bf*)kp, *(const v8bf*)(kp + 16));
      }
    v8f s[2];
    s[0] = zero8(); s[1] = zero8();
#pragma unroll
    for (int kt = 0; kt < 2; ++kt)
#pragma unroll
      for (int dt = 0; dt < 2; ++dt)
        s[kt] = __builtin_amdgcn_wmma_f32_16x16x32_bf16(
            false, A[dt], false, Bk[kt][dt], (short)0, s[kt], false, false);

    // online softmax: row max via fused DPP, raw v_exp_f32
    v8f mb, mn, alpha, rs;
#pragma unroll
    for (int i = 0; i < 8; ++i)
      mb[i] = rowmax16(fmax_raw(s[0][i], s[1][i]));
#pragma unroll
    for (int i = 0; i < 8; ++i) {
      mn[i]    = fmax_raw(mrow[i], mb[i]);
      alpha[i] = exp2_raw(mrow[i] - mn[i]);
      rs[i]    = 0.0f;
    }
#pragma unroll
    for (int kt = 0; kt < 2; ++kt)
#pragma unroll
      for (int i = 0; i < 8; ++i) {
        const float p = exp2_raw(s[kt][i] - mn[i]);
        s[kt][i] = p;
        rs[i] += p;
      }
#pragma unroll
    for (int i = 0; i < 8; ++i) {
      lrow[i] = lrow[i] * alpha[i] + rowsum16(rs[i]);
      mrow[i] = mn[i];
    }
#pragma unroll
    for (int f = 0; f < 4; ++f)
#pragma unroll
      for (int i = 0; i < 8; ++i)
        o[f][i] *= alpha[i];

    // P: C-layout -> LDS -> A-fragment (same wave, DS in-order)
    __bf16* pl = lds_p[wave];
#pragma unroll
    for (int kt = 0; kt < 2; ++kt)
#pragma unroll
      for (int i = 0; i < 8; ++i)
        pl[(half * 8 + i) * 32 + kt * 16 + l15] = (__bf16)s[kt][i];
    const __bf16* pr = pl + l15 * 32 + half * 8;
    v16bf P = cat16(*(const v8bf*)pr, *(const v8bf*)(pr + 16));

    // O(16x64) += P(16x32) . V(32x64) from LDS V tile
    v16bf Bv[4];
#pragma unroll
    for (int nt = 0; nt < 4; ++nt) {
      const __bf16* vp = &lds_v[cur][nt * 16 + l15][half * 8];
      Bv[nt] = cat16(*(const v8bf*)vp, *(const v8bf*)(vp + 16));
    }
#pragma unroll
    for (int nt = 0; nt < 4; ++nt)
      o[nt] = __builtin_amdgcn_wmma_f32_16x16x32_bf16(
          false, P, false, Bv[nt], (short)0, o[nt], false, false);

    __syncthreads();                       // tile fully consumed before reuse
    cur ^= 1;
  }

  // normalize and write ctx (merged-head natural layout)
  v8f inv;
#pragma unroll
  for (int i = 0; i < 8; ++i) inv[i] = 1.0f / lrow[i];
#pragma unroll
  for (int nt = 0; nt < 4; ++nt)
#pragma unroll
    for (int i = 0; i < 8; ++i)
      Cb[(q0 + half * 8 + i) * DM + nt * 16 + l15] = (__bf16)(o[nt][i] * inv[i]);
}

// ---------------------------------------------------------------------------
extern "C" void kernel_launch(void* const* d_in, const int* in_sizes, int n_in,
                              void* d_out, int out_size, void* d_ws, size_t ws_size,
                              hipStream_t stream) {
  const float* query = (const float*)d_in[0];
  const float* key_  = (const float*)d_in[1];
  const float* value = (const float*)d_in[2];
  const float* w_q   = (const float*)d_in[3];
  const float* b_q   = (const float*)d_in[4];
  const float* w_k   = (const float*)d_in[5];
  const float* b_k   = (const float*)d_in[6];
  const float* w_v   = (const float*)d_in[7];
  const float* b_v   = (const float*)d_in[8];
  const float* w_o   = (const float*)d_in[9];
  const float* b_o   = (const float*)d_in[10];
  float* out = (float*)d_out;

  // ws carve (bf16): 4 weights (1M) + xq/xk/xv (8M) + Q/K/V^T (8M) + ctx (8M)
  const size_t WSZ = (size_t)DM * DM;
  const size_t XSZ = (size_t)M_ * DM;
  __bf16* wq = (__bf16*)d_ws;
  __bf16* wk = wq + WSZ;
  __bf16* wv = wk + WSZ;
  __bf16* wo = wv + WSZ;
  __bf16* xq = wo + WSZ;
  __bf16* xk = xq + XSZ;
  __bf16* xv = xk + XSZ;
  __bf16* qp = xv + XSZ;
  __bf16* kp = qp + XSZ;
  __bf16* vt = kp + XSZ;
  __bf16* cx = vt + XSZ;

  const int w4 = (int)(WSZ / 4);
  cvt_f32_to_bf16<<<w4 / 256, 256, 0, stream>>>(w_q, wq, w4);
  cvt_f32_to_bf16<<<w4 / 256, 256, 0, stream>>>(w_k, wk, w4);
  cvt_f32_to_bf16<<<w4 / 256, 256, 0, stream>>>(w_v, wv, w4);
  cvt_f32_to_bf16<<<w4 / 256, 256, 0, stream>>>(w_o, wo, w4);
  const int x4 = (int)(XSZ / 4);
  cvt_f32_to_bf16<<<x4 / 256, 256, 0, stream>>>(query, xq, x4);
  cvt_f32_to_bf16<<<x4 / 256, 256, 0, stream>>>(key_,  xk, x4);
  cvt_f32_to_bf16<<<x4 / 256, 256, 0, stream>>>(value, xv, x4);

  const float QSCALE = 0.125f * 1.44269504088896340736f;  // (1/sqrt(dk))*log2(e)

  dim3 g(M_ / 64, DM / 64);   // 128 x 16 blocks, 128 threads
  gemm16<0><<<g, 128, 0, stream>>>(xq, wq, b_q, qp, QSCALE);  // Q (exp2 domain)
  gemm16<0><<<g, 128, 0, stream>>>(xk, wk, b_k, kp, 1.0f);    // K natural
  gemm16<1><<<g, 128, 0, stream>>>(xv, wv, b_v, vt, 1.0f);    // V transposed

  flash_attn<<<B_ * H_ * (S_ / 64), 128, 0, stream>>>(qp, kp, vt, cx);

  gemm16<2><<<g, 128, 0, stream>>>(cx, wo, b_o, out, 1.0f);   // fp32 output
}
